// IntraConsistencyLoss_26525718020797
// MI455X (gfx1250) — matrix operand order repaired and checked by reading, not compiled, not run on hardware
//
#include <hip/hip_runtime.h>
#include <hip/hip_bf16.h>

typedef __attribute__((ext_vector_type(16))) _Float16 v16h;
typedef __attribute__((ext_vector_type(2)))  _Float16 h2;
typedef __attribute__((ext_vector_type(8)))  float    v8f;
typedef __attribute__((ext_vector_type(8)))  unsigned v8u;
typedef __attribute__((ext_vector_type(4)))  unsigned v4u;

#define ICL_W 512
#define ICL_N 256
#define KTILES (ICL_W / 32)   // 16

// One block (256 threads = 8 wave32) per row n. For each of the 3 heads:
//   Y = D x B, D[i][j] = |h_i - h_j| (built on the fly as 16x32 f16 A-tiles),
//   B columns: [g, 1-g, 1, 0...] precomputed in LDS in WMMA lane layout.
//   Dpp = g^T Y[:,0], Dnn = q^T Y[:,1], Dall = 1^T Y[:,2] via
//   v_wmma_f32_16x16x32_f16; normalizers are closed-form in P = sum(g).
__global__ __launch_bounds__(256) void IntraConsistencyLoss_kernel(
    const float* __restrict__ a_gt,  const float* __restrict__ s_gt,
    const float* __restrict__ e_gt,  const float* __restrict__ a_heat,
    const float* __restrict__ s_heat,const float* __restrict__ e_heat,
    float* __restrict__ out)
{
    __shared__ float sh_h[ICL_W];
    __shared__ float sh_g[ICL_W];
    __shared__ __align__(16) unsigned sh_B[KTILES * 32 * 8];  // packed-f16 B tiles
    __shared__ float sh_acc[3];   // Dpp, Dnn, Dall
    __shared__ float sh_loss;

    const int n    = blockIdx.x;
    const int tid  = threadIdx.x;
    const int lane = tid & 31;
    const int wave = tid >> 5;            // 0..7

    const float* gts[3]   = {a_gt, s_gt, e_gt};
    const float* heats[3] = {a_heat, s_heat, e_heat};

    if (tid == 0) sh_loss = 0.0f;

    const int m      = lane & 15;                  // A row within tile / C col
    const int ncol   = lane & 15;                  // B col = C col
    const int kbaseA = (lane < 16) ? 0 : 16;       // A K-group per half-wave
    const int rbase  = (lane < 16) ? 0 : 8;        // C row base per half-wave

    for (int head = 0; head < 3; ++head) {
        const float* hp = heats[head] + n * ICL_W;
        const float* gp = gts[head]   + n * ICL_W;

        // ---- stage h, g rows into LDS ----
        for (int i = tid; i < ICL_W; i += 256) {
            sh_h[i] = hp[i];
            sh_g[i] = gp[i];
        }

        // ---- precompute packed-f16 B in WMMA lane layout ----
        // dword idx = (t*32 + lane_l)*8 + v ; lane_l holds col N=lane_l%16,
        // K-halfwave base 0/16, two K per dword.
        for (int idx = tid; idx < KTILES * 32 * 8; idx += 256) {
            const int t      = idx >> 8;
            const int lane_l = (idx >> 3) & 31;
            const int v      = idx & 7;
            const int nc     = lane_l & 15;
            const int kb     = (lane_l & 16) ? 16 : 0;
            const int k      = t * 32 + kb + 2 * v;
            const float g0 = gp[k], g1 = gp[k + 1];
            const float b0 = (nc == 0) ? g0 : (nc == 1) ? (1.0f - g0)
                           : (nc == 2) ? 1.0f : 0.0f;
            const float b1 = (nc == 0) ? g1 : (nc == 1) ? (1.0f - g1)
                           : (nc == 2) ? 1.0f : 0.0f;
            h2 pb;
            pb.x = (_Float16)b0;
            pb.y = (_Float16)b1;
            sh_B[idx] = __builtin_bit_cast(unsigned, pb);
        }
        if (tid < 3) sh_acc[tid] = 0.0f;
        __syncthreads();

        // ---- WMMA phase: each wave owns M-tiles {wave + 8t}, t=0..3 ----
        v8f C0 = {}, C1 = {}, C2 = {}, C3 = {};
        float hrow[4];
        #pragma unroll
        for (int t = 0; t < 4; ++t)
            hrow[t] = sh_h[(wave + 8 * t) * 16 + m];

        for (int kt = 0; kt < KTILES; ++kt) {
            const int k0 = kt * 32;
            // B tile: two ds_load_b128 in the per-lane WMMA layout
            const int boff = (kt * 32 + lane) * 8;
            const v4u blo = *(const v4u*)&sh_B[boff];
            const v4u bhi = *(const v4u*)&sh_B[boff + 4];
            v8u bu;
            #pragma unroll
            for (int i = 0; i < 4; ++i) { bu[i] = blo[i]; bu[i + 4] = bhi[i]; }
            const v16h B = __builtin_bit_cast(v16h, bu);

            // 4 M-tiles share this B
            #pragma unroll
            for (int t = 0; t < 4; ++t) {
                const float hm = hrow[t];
                v8u au;
                #pragma unroll
                for (int v = 0; v < 8; ++v) {
                    const int k = k0 + kbaseA + 2 * v;
                    h2 p;
                    p.x = (_Float16)(hm - sh_h[k]);
                    p.y = (_Float16)(hm - sh_h[k + 1]);
                    // |x| in f16 == f16 of |x| (RNE is sign-symmetric):
                    // clear both packed sign bits with one v_and_b32.
                    au[v] = __builtin_bit_cast(unsigned, p) & 0x7FFF7FFFu;
                }
                const v16h A = __builtin_bit_cast(v16h, au);
                if (t == 0) C0 = __builtin_amdgcn_wmma_f32_16x16x32_f16(false, A, false, B, (short)0, C0, false, false);
                if (t == 1) C1 = __builtin_amdgcn_wmma_f32_16x16x32_f16(false, A, false, B, (short)0, C1, false, false);
                if (t == 2) C2 = __builtin_amdgcn_wmma_f32_16x16x32_f16(false, A, false, B, (short)0, C2, false, false);
                if (t == 3) C3 = __builtin_amdgcn_wmma_f32_16x16x32_f16(false, A, false, B, (short)0, C3, false, false);
            }
        }

        // ---- weighted column reduction of the C tiles ----
        if (ncol < 3) {
            float s = 0.0f;
            #pragma unroll
            for (int t = 0; t < 4; ++t) {
                const int i0 = (wave + 8 * t) * 16;
                const v8f& C = (t == 0) ? C0 : (t == 1) ? C1 : (t == 2) ? C2 : C3;
                #pragma unroll
                for (int r = 0; r < 8; ++r) {
                    const float gv = sh_g[i0 + rbase + r];
                    const float w  = (ncol == 0) ? gv
                                   : (ncol == 1) ? (1.0f - gv) : 1.0f;
                    s += w * C[r];
                }
            }
            atomicAdd(&sh_acc[ncol], s);   // ds_add_f32
        }
        __syncthreads();

        // ---- closed-form normalizers ----
        if (tid == 0) {
            float P = 0.0f;
            for (int i = 0; i < ICL_W; ++i) P += sh_g[i];
            const float Q    = (float)ICL_W - P;
            const float Dpp  = sh_acc[0];
            const float Dnn  = sh_acc[1];
            const float Dall = sh_acc[2];
            const float Dmx  = Dall - Dpp - Dnn;
            const float p1 = P * P - P + 1.0f;
            const float p2 = 2.0f * P * Q + 1.0f;
            const float p3 = Q * Q - Q + 1.0f;
            sh_loss += Dpp / p1 + (1.0f - Dmx / p2) + Dnn / p3;
        }
        __syncthreads();   // also guards LDS reuse by next head
    }

    if (tid == 0) out[n] = sh_loss;   // ALPHA == 1.0
}

extern "C" void kernel_launch(void* const* d_in, const int* in_sizes, int n_in,
                              void* d_out, int out_size, void* d_ws, size_t ws_size,
                              hipStream_t stream) {
    const float* a_gt   = (const float*)d_in[0];
    const float* s_gt   = (const float*)d_in[1];
    const float* e_gt   = (const float*)d_in[2];
    const float* a_heat = (const float*)d_in[3];
    const float* s_heat = (const float*)d_in[4];
    const float* e_heat = (const float*)d_in[5];
    float* out = (float*)d_out;

    IntraConsistencyLoss_kernel<<<ICL_N, 256, 0, stream>>>(
        a_gt, s_gt, e_gt, a_heat, s_heat, e_heat, out);
}
